// LPLRQuantizedLinear_37211596652926
// MI455X (gfx1250) — compile-verified, matrix operand order checked by reading, not run on hardware
//
#include <hip/hip_runtime.h>

// ---------------------------------------------------------------------------
// LPLR quantized linear (QuIP#-style E8P + LoRA) for MI455X (gfx1250).
//   h = FWHT(x*SU)                       (LDS butterfly, fp32)
//   Weff = 0.02*grid[Q_idxs] + L@R      (decode + folded low-rank, bf16)
//   O = h @ Weff.T                      (bf16 WMMA, f32 accum, TDM tile DMA)
//   out = FWHT(O) * SV                  (LDS butterfly, fp32)
// ---------------------------------------------------------------------------

typedef __bf16 v8bf  __attribute__((ext_vector_type(8)));
typedef __bf16 v16bf __attribute__((ext_vector_type(16)));
typedef float  v8f   __attribute__((ext_vector_type(8)));
typedef unsigned int u32x4 __attribute__((ext_vector_type(4)));
typedef int          i32x4 __attribute__((ext_vector_type(4)));
typedef int          i32x8 __attribute__((ext_vector_type(8)));

#define N_DIM  4096
#define M_DIM  4096
#define B_ROWS 8192
#define FT     16
#define RES    48
#define QSCALE 0.02f

#define BM 128
#define BN 128
#define BK 64

#if defined(__has_builtin)
#if __has_builtin(__builtin_amdgcn_tensor_load_to_lds) && \
    __has_builtin(__builtin_amdgcn_s_wait_tensorcnt)
#define USE_TDM 1
#endif
#endif
#ifndef USE_TDM
#define USE_TDM 0
#endif

__device__ __forceinline__ unsigned short f2bf(float f) {
  unsigned int u = __float_as_uint(f);
  u += 0x7FFFu + ((u >> 16) & 1u);            // round-to-nearest-even
  return (unsigned short)(u >> 16);
}

#if USE_TDM
// Issue a TDM 2D tile load: rows x BK bf16 elements from a row-major
// [tensorRows][N_DIM] bf16 tensor into LDS (row-major [rows][BK]).
// D# layout per CDNA5 ISA 8.3/8.4 (group0 128b, group1 256b; groups 2/3
// zero => 2D tensor). 6-arg builtin form (clang-23 / therock-10.0 headers).
__device__ __forceinline__ void tdm_load_tile(unsigned lds_addr,
                                              unsigned long long gaddr,
                                              int rows, int tensorRows) {
  u32x4 g0;
  g0.x = 1u;                                        // count=1 (valid D#)
  g0.y = lds_addr;                                  // LDS byte address
  g0.z = (unsigned)(gaddr & 0xFFFFFFFFull);         // global_addr[31:0]
  g0.w = (unsigned)((gaddr >> 32) & 0x01FFFFFFull)  // global_addr[56:32]
         | (2u << 30);                              // type=2 ("image")
  i32x8 g1;
  g1[0] = (1 << 16);                                // data_size=1 (2 bytes)
  g1[1] = (int)((N_DIM & 0xFFFF) << 16);            // tensor_dim0[15:0]
  g1[2] = (int)((N_DIM >> 16) & 0xFFFF) |
          (int)((tensorRows & 0xFFFF) << 16);       // dim0[31:16]|dim1[15:0]
  g1[3] = (int)((tensorRows >> 16) & 0xFFFF) |
          (BK << 16);                               // dim1[31:16]|tile_dim0
  g1[4] = rows;                                     // tile_dim1 (tile_dim2=0)
  g1[5] = N_DIM;                                    // tensor_dim0_stride[31:0]
  g1[6] = 0;                                        // stride0[47:32]|stride1 lo
  g1[7] = 0;
  i32x4 zz4;
  zz4[0] = 0; zz4[1] = 0; zz4[2] = 0; zz4[3] = 0;
  i32x8 zz8;
  zz8[0] = 0; zz8[1] = 0; zz8[2] = 0; zz8[3] = 0;
  zz8[4] = 0; zz8[5] = 0; zz8[6] = 0; zz8[7] = 0;
  __builtin_amdgcn_tensor_load_to_lds(g0, g1, zz4, zz4, zz8, 0);
}
#endif

// ---------------------------------------------------------------------------
// Kernel 1: decode E8P codebook and fold low-rank terms into bf16 Weff[M][N].
//   Weff[m][n] = 0.02 * grid[Q_idxs[m][n/8]][n%8]
//              + sum_r L_ft[m][r]*R_ft[r][n] + sum_r L_res[m][r]*R_res[r][n]
// ---------------------------------------------------------------------------
__global__ void __launch_bounds__(256) decode_weff_kernel(
    const float* __restrict__ grid8, const int* __restrict__ qidx,
    const float* __restrict__ Lft,  const float* __restrict__ Rft,
    const float* __restrict__ Lres, const float* __restrict__ Rres,
    unsigned short* __restrict__ Weff) {
  __shared__ float Lsh[64];
  __shared__ float Rsh[32][256];
  const int t = threadIdx.x;
  const int m = blockIdx.y;
  const int n = blockIdx.x * 256 + t;

  if (t < FT)            Lsh[t] = Lft[m * FT + t];
  else if (t < FT + RES) Lsh[t] = Lres[m * RES + (t - FT)];

  float acc = 0.0f;
  #pragma unroll
  for (int half = 0; half < 2; ++half) {
    __syncthreads();
    #pragma unroll 4
    for (int r = 0; r < 32; ++r) {
      const int g = half * 32 + r;   // row 0..63 of stacked [R_ft; R_res]
      Rsh[r][t] = (g < FT) ? Rft[g * N_DIM + n] : Rres[(g - FT) * N_DIM + n];
    }
    __syncthreads();
    #pragma unroll 8
    for (int r = 0; r < 32; ++r) acc += Lsh[half * 32 + r] * Rsh[r][t];
  }

  const int cw = qidx[m * (N_DIM / 8) + (n >> 3)];
  const float w = grid8[cw * 8 + (n & 7)] * QSCALE + acc;
  Weff[(size_t)m * N_DIM + n] = f2bf(w);
}

// ---------------------------------------------------------------------------
// Kernel 2: h = FWHT_4096(x * SU) / 64, stored as bf16. One row per block.
// ---------------------------------------------------------------------------
__global__ void __launch_bounds__(256) fwht_su_kernel(
    const float* __restrict__ x, const float* __restrict__ SU,
    unsigned short* __restrict__ hout) {
  __shared__ float s[N_DIM];
  const int t = threadIdx.x;
  const size_t row = blockIdx.x;
  const float* xr = x + row * N_DIM;

  #pragma unroll
  for (int k = 0; k < N_DIM / 256; ++k) {
    const int i = t + k * 256;
    s[i] = xr[i] * SU[i];
  }
  __syncthreads();

  for (int hstep = 1; hstep < N_DIM; hstep <<= 1) {
    #pragma unroll
    for (int k = 0; k < (N_DIM / 2) / 256; ++k) {
      const int p = t + k * 256;
      const int j = ((p & ~(hstep - 1)) << 1) | (p & (hstep - 1));
      const float a = s[j], b = s[j + hstep];
      s[j] = a + b;
      s[j + hstep] = a - b;
    }
    __syncthreads();
  }

  const float sc = 1.0f / 64.0f;   // 1/sqrt(4096)
  unsigned short* hr = hout + row * N_DIM;
  #pragma unroll
  for (int k = 0; k < N_DIM / 256; ++k) {
    const int i = t + k * 256;
    hr[i] = f2bf(s[i] * sc);
  }
}

// ---------------------------------------------------------------------------
// Kernel 3: O[b][m] = sum_k h[b][k] * Weff[m][k]   (bf16 WMMA, f32 accum)
// 256 threads (8 waves). Block tile 128x128, K-tile 64, LDS double buffer.
// TDM path: wave 0 DMAs next A/B tiles while all waves run WMMAs.
// ---------------------------------------------------------------------------
__global__ void __launch_bounds__(256) gemm_wmma_kernel(
    const unsigned short* __restrict__ H,   // [B_ROWS][N_DIM] bf16 bits
    const unsigned short* __restrict__ W,   // [M_DIM][N_DIM] bf16 bits
    float* __restrict__ C) {                // [B_ROWS][M_DIM] f32
  __shared__ unsigned short Ash[2][BM * BK];
  __shared__ unsigned short Bsh[2][BN * BK];

  const int t    = threadIdx.x;
  const int lane = t & 31;
  const int wave = t >> 5;
  const int wr   = wave >> 2;     // 0..1 -> 64 batch rows
  const int wc   = wave & 3;      // 0..3 -> 32 output cols
  const int rowBase = blockIdx.y * BM;   // batch rows
  const int colBase = blockIdx.x * BN;   // output cols == Weff rows

  // ISA 16-bit A/B per-lane K mapping: lanes 0-15 hold K {0..7,16..23},
  // lanes 16-31 hold K {8..15,24..31}; M/N index = lane%16.
  const int lsel = (lane < 16) ? 0 : 8;
  const int mrow = lane & 15;

  v8f acc[4][2];
  #pragma unroll
  for (int i = 0; i < 4; ++i)
    #pragma unroll
    for (int j = 0; j < 2; ++j)
      #pragma unroll
      for (int e = 0; e < 8; ++e) acc[i][j][e] = 0.0f;

#if USE_TDM
  const unsigned ldsA0 = (unsigned)(size_t)(&Ash[0][0]);
  const unsigned ldsA1 = (unsigned)(size_t)(&Ash[1][0]);
  const unsigned ldsB0 = (unsigned)(size_t)(&Bsh[0][0]);
  const unsigned ldsB1 = (unsigned)(size_t)(&Bsh[1][0]);
  const unsigned long long gA =
      (unsigned long long)(size_t)(H + (size_t)rowBase * N_DIM);
  const unsigned long long gB =
      (unsigned long long)(size_t)(W + (size_t)colBase * N_DIM);

  if (wave == 0) {   // prologue: DMA tile 0 into buffer 0
    tdm_load_tile(ldsA0, gA, BM, B_ROWS);
    tdm_load_tile(ldsB0, gB, BN, M_DIM);
    __builtin_amdgcn_s_wait_tensorcnt(0);
  }
  __syncthreads();
#endif

  for (int k0 = 0; k0 < N_DIM; k0 += BK) {
#if USE_TDM
    const int cur = (k0 / BK) & 1;
    if ((k0 + BK < N_DIM) && wave == 0) {   // DMA next tile into other buffer
      const unsigned long long off = (unsigned long long)(k0 + BK) * 2ull;
      tdm_load_tile(cur ? ldsA0 : ldsA1, gA + off, BM, B_ROWS);
      tdm_load_tile(cur ? ldsB0 : ldsB1, gB + off, BN, M_DIM);
    }
    const unsigned short* __restrict__ Acur = Ash[cur];
    const unsigned short* __restrict__ Bcur = Bsh[cur];
#else
    // Cooperative fill: 128x64 bf16 per buffer = 1024 16B chunks; 4/thread.
    #pragma unroll
    for (int c = 0; c < 4; ++c) {
      const int chunk = t + c * 256;
      const int r  = chunk >> 3;          // 0..127
      const int kc = (chunk & 7) * 8;     // 0..56 step 8
      *(uint4*)&Ash[0][r * BK + kc] =
          *(const uint4*)&H[(size_t)(rowBase + r) * N_DIM + k0 + kc];
      *(uint4*)&Bsh[0][r * BK + kc] =
          *(const uint4*)&W[(size_t)(colBase + r) * N_DIM + k0 + kc];
    }
    if (k0 + BK < N_DIM) {
      __builtin_prefetch(&H[(size_t)(rowBase + (t >> 3)) * N_DIM + k0 + BK], 0, 1);
      __builtin_prefetch(&W[(size_t)(colBase + (t >> 3)) * N_DIM + k0 + BK], 0, 1);
    }
    __syncthreads();
    const unsigned short* __restrict__ Acur = Ash[0];
    const unsigned short* __restrict__ Bcur = Bsh[0];
#endif

    #pragma unroll
    for (int kk = 0; kk < BK; kk += 32) {
      v16bf afrag[4];
      #pragma unroll
      for (int i = 0; i < 4; ++i) {
        const int r = wr * 64 + i * 16 + mrow;
        const v8bf lo = *(const v8bf*)&Acur[r * BK + kk + lsel];
        const v8bf hi = *(const v8bf*)&Acur[r * BK + kk + 16 + lsel];
        afrag[i] = __builtin_shufflevector(lo, hi, 0, 1, 2, 3, 4, 5, 6, 7,
                                           8, 9, 10, 11, 12, 13, 14, 15);
      }
      v16bf bfrag[2];
      #pragma unroll
      for (int j = 0; j < 2; ++j) {
        const int r = wc * 32 + j * 16 + mrow;
        const v8bf lo = *(const v8bf*)&Bcur[r * BK + kk + lsel];
        const v8bf hi = *(const v8bf*)&Bcur[r * BK + kk + 16 + lsel];
        bfrag[j] = __builtin_shufflevector(lo, hi, 0, 1, 2, 3, 4, 5, 6, 7,
                                           8, 9, 10, 11, 12, 13, 14, 15);
      }
      #pragma unroll
      for (int i = 0; i < 4; ++i)
        #pragma unroll
        for (int j = 0; j < 2; ++j)
          acc[i][j] = __builtin_amdgcn_wmma_f32_16x16x32_bf16(
              false, afrag[i], false, bfrag[j], (short)0, acc[i][j],
              false, false);
    }

#if USE_TDM
    if (wave == 0) __builtin_amdgcn_s_wait_tensorcnt(0);
    __syncthreads();
#else
    __syncthreads();
#endif
  }

  // Epilogue. 16x16 f32 C/D layout: VGPR v, lanes 0-15 -> M=v, N=lane;
  // lanes 16-31 -> M=v+8, N=lane-16.
  const int mq = (lane < 16) ? 0 : 8;
  #pragma unroll
  for (int i = 0; i < 4; ++i)
    #pragma unroll
    for (int j = 0; j < 2; ++j) {
      const int gcol = colBase + wc * 32 + j * 16 + mrow;
      #pragma unroll
      for (int v = 0; v < 8; ++v) {
        const int grow = rowBase + wr * 64 + i * 16 + v + mq;
        C[(size_t)grow * M_DIM + gcol] = acc[i][j][v];
      }
    }
}

// ---------------------------------------------------------------------------
// Kernel 4: out = (FWHT_4096(O) / 64) * SV, fp32 result. One row per block.
// ---------------------------------------------------------------------------
__global__ void __launch_bounds__(256) fwht_sv_kernel(
    const float* __restrict__ O, const float* __restrict__ SV,
    float* __restrict__ outp) {
  __shared__ float s[M_DIM];
  const int t = threadIdx.x;
  const size_t row = blockIdx.x;
  const float* orow = O + row * M_DIM;

  #pragma unroll
  for (int k = 0; k < M_DIM / 256; ++k) {
    const int i = t + k * 256;
    s[i] = orow[i];
  }
  __syncthreads();

  for (int hstep = 1; hstep < M_DIM; hstep <<= 1) {
    #pragma unroll
    for (int k = 0; k < (M_DIM / 2) / 256; ++k) {
      const int p = t + k * 256;
      const int j = ((p & ~(hstep - 1)) << 1) | (p & (hstep - 1));
      const float a = s[j], b = s[j + hstep];
      s[j] = a + b;
      s[j + hstep] = a - b;
    }
    __syncthreads();
  }

  const float sc = 1.0f / 64.0f;
  float* outr = outp + row * M_DIM;
  #pragma unroll
  for (int k = 0; k < M_DIM / 256; ++k) {
    const int i = t + k * 256;
    outr[i] = s[i] * sc * SV[i];
  }
}

// ---------------------------------------------------------------------------
extern "C" void kernel_launch(void* const* d_in, const int* in_sizes, int n_in,
                              void* d_out, int out_size, void* d_ws,
                              size_t ws_size, hipStream_t stream) {
  // setup_inputs order:
  // 0:x 1:SU 2:SV 3:grid 4:L_ft 5:R_ft 6:L_res 7:R_res 8:had_left 9:had_right 10:Q_idxs
  const float* x     = (const float*)d_in[0];
  const float* SU    = (const float*)d_in[1];
  const float* SV    = (const float*)d_in[2];
  const float* grid8 = (const float*)d_in[3];
  const float* Lft   = (const float*)d_in[4];
  const float* Rft   = (const float*)d_in[5];
  const float* Lres  = (const float*)d_in[6];
  const float* Rres  = (const float*)d_in[7];
  const int*   qidx  = (const int*)d_in[10];
  float* outp = (float*)d_out;

  // Workspace: Weff bf16 (32MB) | h bf16 (64MB) | O f32 (128MB)
  char* ws = (char*)d_ws;
  unsigned short* Weff = (unsigned short*)ws;
  unsigned short* Hbuf =
      (unsigned short*)(ws + (size_t)M_DIM * N_DIM * sizeof(unsigned short));
  float* Obuf = (float*)(ws + (size_t)M_DIM * N_DIM * sizeof(unsigned short) +
                         (size_t)B_ROWS * N_DIM * sizeof(unsigned short));

  decode_weff_kernel<<<dim3(N_DIM / 256, M_DIM), 256, 0, stream>>>(
      grid8, qidx, Lft, Rft, Lres, Rres, Weff);
  fwht_su_kernel<<<B_ROWS, 256, 0, stream>>>(x, SU, Hbuf);
  gemm_wmma_kernel<<<dim3(M_DIM / BN, B_ROWS / BM), 256, 0, stream>>>(
      Hbuf, Weff, Obuf);
  fwht_sv_kernel<<<B_ROWS, 256, 0, stream>>>(Obuf, SV, outp);
}
